// RAFTCore32_29686813950756
// MI455X (gfx1250) — matrix operand order repaired and checked by reading, not compiled, not run on hardware
//
#include <hip/hip_runtime.h>
#include <hip/hip_bf16.h>
#include <math.h>

// ---------------------------------------------------------------------------
// MI455X (gfx1250) RAFT forward. All conv / correlation GEMMs run on
// v_wmma_f32_16x16x32_bf16 (wave32). Conv uses a 32-pixel M tile (2 WMMAs per
// staged B fragment). The correlation GEMM stages its B tiles with
// global_load_async_to_lds_b32 (ASYNCcnt) — the CDNA5 async LDS-copy path.
// ---------------------------------------------------------------------------

typedef __bf16 bf16;
typedef __attribute__((ext_vector_type(16))) __bf16 v16bf;
typedef __attribute__((ext_vector_type(8)))  float  v8f;

#define BATCH 16
#define CCH   128

__device__ __forceinline__ float sigmoidf_(float x) { return 1.f / (1.f + __expf(-x)); }

// --------------------------- weight pre-pack -------------------------------
// dst[((kp*Nt + nt)*32 + lane)*16 + e] = W[k][n], n = nt*16 + lane%16,
// k = kp*32 + e + 16*(lane/16)   (B-fragment layout for 16x16x32 bf16 WMMA)
__global__ void pack_w_kernel(const float* __restrict__ w, bf16* __restrict__ dst,
                              int K, int Cout, int Kp, int Nt) {
  int total = Kp * Nt * 16;
  for (int t = blockIdx.x * blockDim.x + threadIdx.x; t < total; t += gridDim.x * blockDim.x) {
    int e    = t & 15;
    int lane = (t >> 4) & 31;
    int kt   = t >> 9;        // kp*Nt + nt
    int nt   = kt % Nt;
    int kp   = kt / Nt;
    int n = nt * 16 + (lane & 15);
    int k = kp * 32 + e + ((lane >> 4) << 4);
    float v = (k < K && n < Cout) ? w[(size_t)n * K + k] : 0.f;
    dst[t] = (bf16)v;
  }
}

// --------------------------- implicit-GEMM conv ----------------------------
// M = B*OH*OW pixels (32-pixel tile/block), N = Cout, K = Cin*KS*KS.
// Block = 8 waves; wave w owns N-tile (blockIdx.y*8 + w). Each wave reuses one
// B fragment across two A fragments -> 2 WMMAs per K step.
template <int KS>
__global__ __launch_bounds__(256) void conv_wmma_kernel(
    const float* __restrict__ in, const bf16* __restrict__ wpk,
    const float* __restrict__ bias, float* __restrict__ out,
    int Cin, int H, int W, int Cout, int OH, int OW,
    int stride, int pad, int Kp, int Nt) {
  __shared__ bf16 Alds[32 * 40];
  const int tid  = threadIdx.x;
  const int wave = tid >> 5, lane = tid & 31;
  const int hi = lane >> 4, lm = lane & 15;
  const int m0   = blockIdx.x * 32;
  const int OHW  = OH * OW;
  const int Mtot = BATCH * OHW;
  const int nt   = blockIdx.y * 8 + wave;
  const int K    = Cin * KS * KS;
  v8f acc0 = {}, acc1 = {};
  for (int k0 = 0; k0 < Kp; k0 += 32) {
    __syncthreads();
    for (int idx = tid; idx < 1024; idx += 256) {
      int p = idx >> 5, kk = idx & 31;
      int k = k0 + kk;
      float v = 0.f;
      int gm = m0 + p;
      if (gm < Mtot && k < K) {
        int b = gm / OHW; int rem = gm - b * OHW;
        int oy = rem / OW, ox = rem - oy * OW;
        int ci, ky, kx;
        if (KS == 1) { ci = k; ky = 0; kx = 0; }
        else { ci = k / 9; int r = k - ci * 9; ky = r / 3; kx = r - ky * 3; }
        int iy = oy * stride + ky - pad;
        int ix = ox * stride + kx - pad;
        if (iy >= 0 && iy < H && ix >= 0 && ix < W)
          v = in[((size_t)(b * Cin + ci) * H + iy) * W + ix];
      }
      Alds[p * 40 + kk] = (bf16)v;
    }
    __syncthreads();
    v16bf a0, a1;
#pragma unroll
    for (int e = 0; e < 16; ++e) {
      int kl = e + (hi << 4);   // contiguous k = 16*hi + e (matches packed B)
      a0[e] = Alds[lm * 40 + kl];
      a1[e] = Alds[(16 + lm) * 40 + kl];
    }
    v16bf bfrag = {};
    if (nt < Nt) {
      const bf16* src = wpk + ((size_t)((k0 >> 5) * Nt + nt) * 32 + lane) * 16;
#pragma unroll
      for (int e = 0; e < 16; ++e) bfrag[e] = src[e];   // contiguous 32B/lane
    }
    acc0 = __builtin_amdgcn_wmma_f32_16x16x32_bf16(false, a0, false, bfrag,
                                                   (short)0, acc0, false, false);
    acc1 = __builtin_amdgcn_wmma_f32_16x16x32_bf16(false, a1, false, bfrag,
                                                   (short)0, acc1, false, false);
  }
  if (nt < Nt) {
    int n = nt * 16 + lm;
    if (n < Cout) {
      float bv = bias ? bias[n] : 0.f;
#pragma unroll
      for (int r = 0; r < 8; ++r) {
        int gm = m0 + r + (hi << 3);
        if (gm < Mtot) {
          int b = gm / OHW; int rem = gm - b * OHW;
          out[(size_t)(b * Cout + n) * OHW + rem] = acc0[r] + bv;
        }
        int gm1 = m0 + 16 + r + (hi << 3);
        if (gm1 < Mtot) {
          int b = gm1 / OHW; int rem = gm1 - b * OHW;
          out[(size_t)(b * Cout + n) * OHW + rem] = acc1[r] + bv;
        }
      }
    }
  }
}

// --------------------------- correlation GEMM ------------------------------
// corr[b][m][n] = sum_c f1[b][c][m] * f2[b][c][n], M=N=1024, K=128, bf16.
// B tiles staged via global_load_async_to_lds_b32 (ASYNCcnt).
__global__ __launch_bounds__(256) void corr_wmma_kernel(
    const bf16* __restrict__ f1, const bf16* __restrict__ f2,
    float* __restrict__ corr, int C, int HW) {
  __shared__ bf16 Alds[16 * 40];
  __shared__ bf16 Blds[8 * 32 * 18];   // row stride 18 halves = 36B (4B aligned)
  const int tid = threadIdx.x;
  const int wave = tid >> 5, lane = tid & 31;
  const int hi = lane >> 4, lm = lane & 15;
  const int m0  = blockIdx.x * 16;
  const int n0p = blockIdx.y * 128;
  const int b   = blockIdx.z;
  const bf16* f1b = f1 + (size_t)b * C * HW;
  const bf16* f2b = f2 + (size_t)b * C * HW;
  v8f acc = {};
  for (int k0 = 0; k0 < C; k0 += 32) {
    __syncthreads();
    // A tile: regular LDS stores (vectorizes into contiguous frag reads later)
    for (int idx = tid; idx < 512; idx += 256) {
      int m = idx & 15, kk = idx >> 4;
      Alds[m * 40 + kk] = f1b[(size_t)(k0 + kk) * HW + m0 + m];
    }
    // B tiles: async copy global->LDS, 4B per op, 2048 ops per K step
    for (int idx = tid; idx < 2048; idx += 256) {
      int np = idx & 7, kk = (idx >> 3) & 31, w = idx >> 8;
      unsigned long long ga =
          (unsigned long long)(f2b + (size_t)(k0 + kk) * HW + n0p + w * 16 + np * 2);
      unsigned la = (unsigned)(unsigned long long)(&Blds[w * 576 + kk * 18 + np * 2]);
      asm volatile("global_load_async_to_lds_b32 %0, %1, off"
                   :: "v"(la), "v"(ga) : "memory");
    }
    asm volatile("s_wait_asynccnt 0x0" ::: "memory");
    __syncthreads();
    v16bf afrag, bfrag;
#pragma unroll
    for (int e = 0; e < 16; ++e) {
      int kl = e + (hi << 4);
      afrag[e] = Alds[lm * 40 + kl];
    }
#pragma unroll
    for (int e = 0; e < 16; ++e)
      bfrag[e] = Blds[wave * 576 + (e + (hi << 4)) * 18 + lm];
    acc = __builtin_amdgcn_wmma_f32_16x16x32_bf16(false, afrag, false, bfrag,
                                                  (short)0, acc, false, false);
  }
  int n = n0p + wave * 16 + lm;
#pragma unroll
  for (int r = 0; r < 8; ++r) {
    int m = m0 + r + (hi << 3);
    corr[((size_t)b * HW + m) * HW + n] = acc[r];
  }
}

// --------------------------- GroupNorm (+SiLU) -----------------------------
__global__ void gn_kernel(const float* __restrict__ in, const float* __restrict__ gam,
                          const float* __restrict__ bet, float* __restrict__ out,
                          int HW, int act) {
  const int b = blockIdx.x >> 4, grp = blockIdx.x & 15;
  const float* x = in  + ((size_t)b * CCH + grp * 8) * HW;
  float*       o = out + ((size_t)b * CCH + grp * 8) * HW;
  const int n = 8 * HW;
  float s = 0.f, s2 = 0.f;
  for (int i = threadIdx.x; i < n; i += 256) { float v = x[i]; s += v; s2 += v * v; }
  __shared__ float rs[256], rq[256];
  rs[threadIdx.x] = s; rq[threadIdx.x] = s2;
  __syncthreads();
  for (int off = 128; off > 0; off >>= 1) {
    if (threadIdx.x < off) { rs[threadIdx.x] += rs[threadIdx.x + off]; rq[threadIdx.x] += rq[threadIdx.x + off]; }
    __syncthreads();
  }
  float mean = rs[0] / n;
  float var  = rq[0] / n - mean * mean;
  float inv  = rsqrtf(var + 1e-5f);
  for (int i = threadIdx.x; i < n; i += 256) {
    int c = grp * 8 + i / HW;
    float v = (x[i] - mean) * inv * gam[c] + bet[c];
    if (act) v = v * sigmoidf_(v);
    o[i] = v;
  }
}

// --------------------------- elementwise helpers ---------------------------
__global__ void unary_kernel(const float* a, float* out, int n, int mode) {
  int t = blockIdx.x * 256 + threadIdx.x; if (t >= n) return;
  float v = a[t];
  if (mode == 1) v = v * sigmoidf_(v);
  else if (mode == 2) v = tanhf(v);
  out[t] = v;
}
__global__ void binop_kernel(const float* a, const float* b, float* out, int n, int mode) {
  int t = blockIdx.x * 256 + threadIdx.x; if (t >= n) return;
  float v = a[t] + b[t];
  if (mode == 1) v = v * sigmoidf_(v);
  out[t] = v;
}
__global__ void fill0_kernel(float* p, int n) {
  int t = blockIdx.x * 256 + threadIdx.x; if (t < n) p[t] = 0.f;
}
__global__ void concat2_kernel(const float* a, const float* b, const float* h,
                               float* out, int HW, int mode, int total) {
  int t = blockIdx.x * 256 + threadIdx.x; if (t >= total) return;
  int p  = t % HW;
  int ch = (t / HW) & 255;
  int bb = t / (HW * 256);
  float v;
  if (ch < 128) v = a[((size_t)bb * 128 + ch) * HW + p];
  else {
    float u = b[((size_t)bb * 128 + ch - 128) * HW + p];
    if (mode == 1) v = sigmoidf_(u) * h[((size_t)bb * 128 + ch - 128) * HW + p];
    else v = u;
  }
  out[t] = v;
}
__global__ void gru_update_kernel(const float* zc, const float* hc, float* h, int n) {
  int t = blockIdx.x * 256 + threadIdx.x; if (t >= n) return;
  float z = sigmoidf_(zc[t]);
  h[t] = (1.f - z) * h[t] + z * tanhf(hc[t]);
}

// --------------------------- resize / pool / l2norm ------------------------
__global__ void resize2x_kernel(const float* __restrict__ in, float* __restrict__ out,
                                int BC, int H) {
  int W = H, OH = 2 * H, OW = 2 * H;
  int total = BC * OH * OW;
  int t = blockIdx.x * 256 + threadIdx.x; if (t >= total) return;
  int ox = t % OW, oy = (t / OW) % OH, bc = t / (OW * OH);
  float sx = (ox + 0.5f) * 0.5f - 0.5f;
  float sy = (oy + 0.5f) * 0.5f - 0.5f;
  int x0f = (int)floorf(sx); float wx = sx - x0f;
  int y0f = (int)floorf(sy); float wy = sy - y0f;
  int x0 = min(W - 1, max(0, x0f)), x1 = min(W - 1, max(0, x0f + 1));
  int y0 = min(H - 1, max(0, y0f)), y1 = min(H - 1, max(0, y0f + 1));
  const float* s = in + (size_t)bc * H * W;
  out[t] = (1.f - wy) * ((1.f - wx) * s[y0 * W + x0] + wx * s[y0 * W + x1]) +
           wy * ((1.f - wx) * s[y1 * W + x0] + wx * s[y1 * W + x1]);
}
__global__ void pool2_kernel(const float* __restrict__ src, float* __restrict__ dst,
                             int N, int h) {
  int oh = h >> 1;
  int total = N * oh * oh;
  int t = blockIdx.x * 256 + threadIdx.x; if (t >= total) return;
  int x = t % oh, y = (t / oh) % oh, n = t / (oh * oh);
  const float* s = src + (size_t)n * h * h;
  dst[t] = 0.25f * (s[(2 * y) * h + 2 * x] + s[(2 * y) * h + 2 * x + 1] +
                    s[(2 * y + 1) * h + 2 * x] + s[(2 * y + 1) * h + 2 * x + 1]);
}
__global__ void l2norm_kernel(const float* __restrict__ f, bf16* __restrict__ out, int HW) {
  int t = blockIdx.x * 256 + threadIdx.x;
  if (t >= BATCH * HW) return;
  int p = t % HW, b = t / HW;
  const float* src = f + (size_t)b * CCH * HW + p;
  float s = 0.f;
  for (int c = 0; c < CCH; ++c) { float v = src[(size_t)c * HW]; s += v * v; }
  float inv = 1.f / fmaxf(sqrtf(s), 1e-6f);
  bf16* o = out + (size_t)b * CCH * HW + p;
  for (int c = 0; c < CCH; ++c) o[(size_t)c * HW] = (bf16)(src[(size_t)c * HW] * inv);
}

// --------------------------- correlation sampling --------------------------
__device__ __forceinline__ float bsample(const float* __restrict__ v, int h, int w,
                                         float x, float y) {
  float x0 = floorf(x), y0 = floorf(y);
  float wx1 = x - x0, wx0 = 1.f - wx1;
  float wy1 = y - y0, wy0 = 1.f - wy1;
  float r = 0.f;
#pragma unroll
  for (int dy = 0; dy < 2; ++dy)
#pragma unroll
    for (int dx = 0; dx < 2; ++dx) {
      float xi = x0 + dx, yi = y0 + dy;
      if (xi >= 0.f && xi <= (float)(w - 1) && yi >= 0.f && yi <= (float)(h - 1)) {
        float wt = (dx ? wx1 : wx0) * (dy ? wy1 : wy0);
        r += wt * v[(int)yi * w + (int)xi];
      }
    }
  return r;
}
__global__ void motion_input_kernel(const float* __restrict__ corr0,
                                    const float* __restrict__ corr1,
                                    const float* __restrict__ corr2,
                                    const float* __restrict__ flow,
                                    float* __restrict__ cf) {
  int t = blockIdx.x * 256 + threadIdx.x;
  if (t >= BATCH * 1024 * 81) return;
  int k   = t % 81;
  int pix = (t / 81) & 1023;
  int b   = t / (81 * 1024);
  int w1 = pix & 31, h1 = pix >> 5;
  float fx = flow[(size_t)(b * 2 + 0) * 1024 + pix];
  float fy = flow[(size_t)(b * 2 + 1) * 1024 + pix];
  float cx = w1 + fx, cy = h1 + fy;
  float dx = (float)(k % 9 - 4), dy = (float)(k / 9 - 4);
  size_t base = (size_t)b * 245 * 1024 + pix;
  cf[base + (size_t)k * 1024]         = bsample(corr0 + ((size_t)b * 1024 + pix) * 1024, 32, 32, cx + dx, cy + dy);
  cf[base + (size_t)(81 + k) * 1024]  = bsample(corr1 + ((size_t)b * 1024 + pix) * 256, 16, 16, cx * 0.5f + dx, cy * 0.5f + dy);
  cf[base + (size_t)(162 + k) * 1024] = bsample(corr2 + ((size_t)b * 1024 + pix) * 64, 8, 8, cx * 0.25f + dx, cy * 0.25f + dy);
  if (k == 0) {
    cf[base + (size_t)243 * 1024] = fx;
    cf[base + (size_t)244 * 1024] = fy;
  }
}

// --------------------------- convex upsampling -----------------------------
__global__ void raft_up_kernel(const float* __restrict__ mask, const float* __restrict__ flow,
                               float* __restrict__ out, int t) {
  int id = blockIdx.x * 256 + threadIdx.x;
  if (id >= BATCH * 4096) return;
  int ox = id & 63, oy = (id >> 6) & 63, b = id >> 12;
  int i = oy >> 1, di = oy & 1, j = ox >> 1, dj = ox & 1;
  float m[9], mx = -1e30f;
#pragma unroll
  for (int k = 0; k < 9; ++k) {
    m[k] = mask[((size_t)(b * 36 + (k * 4 + di * 2 + dj)) << 10) + (i << 5) + j];
    mx = fmaxf(mx, m[k]);
  }
  float se = 0.f;
#pragma unroll
  for (int k = 0; k < 9; ++k) { m[k] = __expf(m[k] - mx); se += m[k]; }
  float inv = 1.f / se;
  float fx = 0.f, fy = 0.f;
#pragma unroll
  for (int k = 0; k < 9; ++k) {
    int yy = i + k / 3 - 1, xx = j + k % 3 - 1;
    if (yy >= 0 && yy < 32 && xx >= 0 && xx < 32) {
      float wt = m[k] * inv;
      fx += wt * flow[(size_t)(b * 2 + 0) * 1024 + (yy << 5) + xx];
      fy += wt * flow[(size_t)(b * 2 + 1) * 1024 + (yy << 5) + xx];
    }
  }
  size_t base = (size_t)(b * 6 + t) * 2 * 4096 + (oy << 6) + ox;
  out[base]        = 2.f * fx;
  out[base + 4096] = 2.f * fy;
}

// ===========================================================================
// Host-side orchestration
// ===========================================================================
namespace {

struct Ctx {
  void* const* din;
  hipStream_t st;
  char* ws;
  size_t cur;
  void* alloc(size_t bytes) {
    size_t a = (cur + 255) & ~(size_t)255;
    cur = a + bytes;
    return ws + a;
  }
  const float* P(int i) const { return (const float*)din[i]; }
};

static inline int ceil_div(int a, int b) { return (a + b - 1) / b; }

static const bf16* packConv(Ctx& c, int idx, int Cin, int Cout, int ks) {
  int K = Cin * ks * ks;
  int Kp = (K + 31) & ~31;
  int Nt = (Cout + 15) >> 4;
  size_t total = (size_t)Kp * Nt * 16;
  bf16* dst = (bf16*)c.alloc(total * sizeof(bf16));
  pack_w_kernel<<<ceil_div((int)total, 256), 256, 0, c.st>>>(c.P(idx), dst, K, Cout, Kp, Nt);
  return dst;
}

static void convk(Ctx& c, const float* in, const bf16* w, int Cin, int H, int Wd, int Cout,
                  float* out, int ks, int stride, int pad, const float* bias) {
  int OH = (H + 2 * pad - ks) / stride + 1;
  int OW = (Wd + 2 * pad - ks) / stride + 1;
  int Mtot = BATCH * OH * OW;
  int Kp = (Cin * ks * ks + 31) & ~31;
  int Nt = (Cout + 15) >> 4;
  dim3 grid(ceil_div(Mtot, 32), ceil_div(Nt, 8));
  if (ks == 3)
    conv_wmma_kernel<3><<<grid, 256, 0, c.st>>>(in, w, bias, out, Cin, H, Wd, Cout, OH, OW, stride, pad, Kp, Nt);
  else
    conv_wmma_kernel<1><<<grid, 256, 0, c.st>>>(in, w, bias, out, Cin, H, Wd, Cout, OH, OW, stride, pad, Kp, Nt);
}

static void gnk(Ctx& c, const float* in, int gi, int bi, float* out, int HW, int act) {
  gn_kernel<<<BATCH * 16, 256, 0, c.st>>>(in, c.P(gi), c.P(bi), out, HW, act);
}
static void unary(Ctx& c, const float* a, float* out, int n, int mode) {
  unary_kernel<<<ceil_div(n, 256), 256, 0, c.st>>>(a, out, n, mode);
}
static void binop(Ctx& c, const float* a, const float* b, float* out, int n, int mode) {
  binop_kernel<<<ceil_div(n, 256), 256, 0, c.st>>>(a, b, out, n, mode);
}
static void concat2(Ctx& c, const float* a, const float* b, const float* h, float* out, int mode) {
  int total = BATCH * 256 * 1024;
  concat2_kernel<<<ceil_div(total, 256), 256, 0, c.st>>>(a, b, h, out, 1024, mode, total);
}

// ---- parameter index tables (top-level dict insertion order, params sorted) ----
struct RBp  { const bf16 *c1, *c2; int g1, b1, g2, b2; };
struct BLKp { const bf16* cw; int g, b; RBp rb; };
struct UNp  { const bf16* stem; int sg, sb, cin; RBp srb; BLKp d1, d2, d3, u2, u1; RBp fuse; };

static RBp mkRB(Ctx& c, int base) {   // keys: b1,b2,c1,c2,g1,g2
  RBp r;
  r.b1 = base; r.b2 = base + 1;
  r.c1 = packConv(c, base + 2, 128, 128, 3);
  r.c2 = packConv(c, base + 3, 128, 128, 3);
  r.g1 = base + 4; r.g2 = base + 5;
  return r;
}
static BLKp mkBLK(Ctx& c, int base) { // keys: b,c,g,rb
  BLKp b;
  b.b = base;
  b.cw = packConv(c, base + 1, 128, 128, 3);
  b.g = base + 2;
  b.rb = mkRB(c, base + 3);
  return b;
}
static UNp mkFeatUN(Ctx& c, int F, int cin) { // keys: d1,d2,d3,fuse,stem_b,stem_c,stem_g,stem_rb,u1,u2
  UNp u; u.cin = cin;
  u.d1 = mkBLK(c, F); u.d2 = mkBLK(c, F + 9); u.d3 = mkBLK(c, F + 18);
  u.fuse = mkRB(c, F + 27);
  u.sb = F + 33; u.stem = packConv(c, F + 34, cin, 128, 3); u.sg = F + 35;
  u.srb = mkRB(c, F + 36);
  u.u1 = mkBLK(c, F + 42); u.u2 = mkBLK(c, F + 51);
  return u;
}
static UNp mkCtxUN(Ctx& c) { // keys: d1,d2,d3,fuse,pc,ph,stem_b,stem_c,stem_g,stem_rb,u1,u2
  UNp u; u.cin = 6;
  u.d1 = mkBLK(c, 3); u.d2 = mkBLK(c, 12); u.d3 = mkBLK(c, 21);
  u.fuse = mkRB(c, 30);
  u.sb = 38; u.stem = packConv(c, 39, 6, 128, 3); u.sg = 40;
  u.srb = mkRB(c, 41);
  u.u1 = mkBLK(c, 47); u.u2 = mkBLK(c, 56);
  return u;
}

static void resblock(Ctx& c, const float* x, const RBp& rb, int H,
                     float* tA, float* tB, float* out) {
  int HW = H * H;
  convk(c, x, rb.c1, 128, H, H, 128, tA, 3, 1, 1, nullptr);
  gnk(c, tA, rb.g1, rb.b1, tB, HW, 1);
  convk(c, tB, rb.c2, 128, H, H, 128, tA, 3, 1, 1, nullptr);
  gnk(c, tA, rb.g2, rb.b2, tB, HW, 0);
  binop(c, x, tB, out, BATCH * CCH * HW, 1);   // silu(x + y)
}
static void runDown(Ctx& c, const float* in, const BLKp& P, int H,
                    float* tA, float* tB, float* tC, float* sOut) {
  int OH = H / 2;
  convk(c, in, P.cw, 128, H, H, 128, tA, 3, 2, 1, nullptr);
  gnk(c, tA, P.g, P.b, tB, OH * OH, 1);
  resblock(c, tB, P.rb, OH, tA, tC, sOut);
}
static void runUp(Ctx& c, const float* in, const BLKp& P, int Hin,
                  float* tA, float* tB, float* tC, float* out) {
  int H = Hin * 2;
  resize2x_kernel<<<ceil_div(BATCH * CCH * H * H, 256), 256, 0, c.st>>>(in, tA, BATCH * CCH, Hin);
  convk(c, tA, P.cw, 128, H, H, 128, tB, 3, 1, 1, nullptr);
  gnk(c, tB, P.g, P.b, tC, H * H, 1);
  resblock(c, tC, P.rb, H, tA, tB, out);
}
static void runUnet(Ctx& c, const float* x, const UNp& P, float* out,
                    float** a64, float** a32, float** a16, float** a8) {
  convk(c, x, P.stem, P.cin, 64, 64, 128, a64[0], 3, 1, 1, nullptr);
  gnk(c, a64[0], P.sg, P.sb, a64[1], 4096, 1);
  resblock(c, a64[1], P.srb, 64, a64[0], a64[2], a64[3]);                 // s64
  runDown(c, a64[3], P.d1, 64, a32[0], a32[1], a32[2], a32[3]);           // s32
  runDown(c, a32[3], P.d2, 32, a16[0], a16[1], a16[2], a16[3]);           // s16
  runDown(c, a16[3], P.d3, 16, a8[0], a8[1], a8[2], a8[3]);               // s8
  runUp(c, a8[3], P.u2, 8, a16[0], a16[1], a16[2], a16[0]);
  binop(c, a16[0], a16[3], a16[1], BATCH * CCH * 256, 0);                 // y16
  runUp(c, a16[1], P.u1, 16, a32[0], a32[1], a32[2], a32[0]);
  binop(c, a32[0], a32[3], a32[1], BATCH * CCH * 1024, 0);                // y32
  resblock(c, a32[1], P.fuse, 32, a32[0], a32[2], out);
}

} // namespace

extern "C" void kernel_launch(void* const* d_in, const int* in_sizes, int n_in,
                              void* d_out, int out_size, void* d_ws, size_t ws_size,
                              hipStream_t stream) {
  (void)in_sizes; (void)n_in; (void)out_size; (void)ws_size;
  Ctx c{d_in, stream, (char*)d_ws, 0};

  // -------- pack all conv weights into WMMA B-fragment layout --------
  UNp Uctx  = mkCtxUN(c);
  const bf16* Wpc = packConv(c, 36, 128, 128, 3);
  const bf16* Wph = packConv(c, 37, 128, 128, 3);
  const bf16* Wcf = packConv(c, 65, 256, 128, 1);          // ctx_fuse (1x1)
  UNp Uf1 = mkFeatUN(c, 66, 3);
  UNp Uf2 = mkFeatUN(c, 126, 3);
  const bf16* Wfh = packConv(c, 186, 128, 2, 3);           // flow_head (+bias 187)
  const bf16* Wgh = packConv(c, 188, 256, 128, 3);
  const bf16* Wgr = packConv(c, 189, 256, 128, 3);
  const bf16* Wgz = packConv(c, 190, 256, 128, 3);
  int motB = 191;
  const bf16* Wmc = packConv(c, 192, 245, 128, 3);
  int motG = 193;
  RBp MotRB = mkRB(c, 194);
  int upB = 200;
  const bf16* Wup1 = packConv(c, 201, 128, 128, 3);
  const bf16* Wup2 = packConv(c, 202, 128, 36, 1);         // (+bias 203)
  int upG = 204;

  // -------- activation scratch --------
  float* a64[4]; for (int i = 0; i < 4; ++i) a64[i] = (float*)c.alloc(33554432);
  float* a32[4]; for (int i = 0; i < 4; ++i) a32[i] = (float*)c.alloc(8388608);
  float* a16[4]; for (int i = 0; i < 4; ++i) a16[i] = (float*)c.alloc(2097152);
  float* a8[4];  for (int i = 0; i < 4; ++i) a8[i]  = (float*)c.alloc(524288);
  float* fo1  = (float*)c.alloc(8388608);
  float* fo2  = (float*)c.alloc(8388608);
  float* yctx = (float*)c.alloc(8388608);
  bf16* f1b = (bf16*)c.alloc(4194304);
  bf16* f2b = (bf16*)c.alloc(4194304);
  float* h32 = (float*)c.alloc(8388608);
  float* c32 = (float*)c.alloc(8388608);
  float* x32 = (float*)c.alloc(8388608);
  float* zc  = (float*)c.alloc(8388608);
  float* rc  = (float*)c.alloc(8388608);
  float* hc  = (float*)c.alloc(8388608);
  float* xh  = (float*)c.alloc(16777216);
  float* xrh = (float*)c.alloc(16777216);
  float* cf  = (float*)c.alloc((size_t)BATCH * 245 * 1024 * 4);
  float* mask = (float*)c.alloc((size_t)BATCH * 36 * 1024 * 4);
  float* flow  = (float*)c.alloc(131072);
  float* delta = (float*)c.alloc(131072);

  float* dout = (float*)d_out;
  const int N32 = BATCH * CCH * 1024;

  // -------- feature + context UNets --------
  runUnet(c, c.P(0), Uf1, fo1, a64, a32, a16, a8);
  l2norm_kernel<<<ceil_div(BATCH * 1024, 256), 256, 0, stream>>>(fo1, f1b, 1024);
  runUnet(c, c.P(1), Uf2, fo2, a64, a32, a16, a8);
  l2norm_kernel<<<ceil_div(BATCH * 1024, 256), 256, 0, stream>>>(fo2, f2b, 1024);
  runUnet(c, c.P(2), Uctx, yctx, a64, a32, a16, a8);

  convk(c, yctx, Wph, 128, 32, 32, 128, a32[0], 3, 1, 1, nullptr);
  unary(c, a32[0], h32, N32, 2);                           // h = tanh(conv)
  convk(c, yctx, Wpc, 128, 32, 32, 128, a32[0], 3, 1, 1, nullptr);
  unary(c, a32[0], c32, N32, 1);                           // c32 = silu(conv)

  // -------- correlation pyramid (overlays the dead 64x64 pool) --------
  float* corr0 = a64[0];                                   // 67MB: a64[0]+a64[1] (contiguous)
  float* corr1 = a64[2];
  float* corr2 = a64[3];
  {
    dim3 grid(64, 8, BATCH);
    corr_wmma_kernel<<<grid, 256, 0, stream>>>(f1b, f2b, corr0, 128, 1024);
  }
  pool2_kernel<<<ceil_div(BATCH * 1024 * 256, 256), 256, 0, stream>>>(corr0, corr1, BATCH * 1024, 32);
  pool2_kernel<<<ceil_div(BATCH * 1024 * 64, 256), 256, 0, stream>>>(corr1, corr2, BATCH * 1024, 16);

  // -------- iterative GRU refinement --------
  fill0_kernel<<<ceil_div(32768, 256), 256, 0, stream>>>(flow, 32768);
  for (int t = 0; t < 6; ++t) {
    motion_input_kernel<<<ceil_div(BATCH * 1024 * 81, 256), 256, 0, stream>>>(corr0, corr1, corr2, flow, cf);
    convk(c, cf, Wmc, 245, 32, 32, 128, a32[0], 3, 1, 1, nullptr);
    gnk(c, a32[0], motG, motB, a32[1], 1024, 1);
    resblock(c, a32[1], MotRB, 32, a32[0], a32[2], a32[3]);        // m = a32[3]
    concat2(c, a32[3], c32, nullptr, xh, 0);
    convk(c, xh, Wcf, 256, 32, 32, 128, x32, 1, 1, 0, nullptr);    // x
    concat2(c, x32, h32, nullptr, xh, 0);                          // [x,h]
    convk(c, xh, Wgz, 256, 32, 32, 128, zc, 3, 1, 1, nullptr);
    convk(c, xh, Wgr, 256, 32, 32, 128, rc, 3, 1, 1, nullptr);
    concat2(c, x32, rc, h32, xrh, 1);                              // [x, sigmoid(rc)*h]
    convk(c, xrh, Wgh, 256, 32, 32, 128, hc, 3, 1, 1, nullptr);
    gru_update_kernel<<<ceil_div(N32, 256), 256, 0, stream>>>(zc, hc, h32, N32);
    convk(c, h32, Wfh, 128, 32, 32, 2, delta, 3, 1, 1, c.P(187));
    binop(c, flow, delta, flow, 32768, 0);                         // coords1 += delta
    convk(c, h32, Wup1, 128, 32, 32, 128, a32[0], 3, 1, 1, nullptr);
    gnk(c, a32[0], upG, upB, a32[1], 1024, 1);
    convk(c, a32[1], Wup2, 128, 32, 32, 36, mask, 1, 1, 0, c.P(203));
    raft_up_kernel<<<ceil_div(BATCH * 4096, 256), 256, 0, stream>>>(mask, flow, dout, t);
  }
  // final low-res flow after the stacked (B,6,2,64,64) flows
  unary(c, flow, dout + (size_t)BATCH * 6 * 2 * 4096, 32768, 0);
}
